// WassersteinOptimalTransport_88304527606085
// MI455X (gfx1250) — matrix operand order repaired and checked by reading, not compiled, not run on hardware
//
#include <hip/hip_runtime.h>

typedef _Float16 h8  __attribute__((ext_vector_type(8)));
typedef __attribute__((ext_vector_type(16))) _Float16 v16h;
typedef __attribute__((ext_vector_type(8)))  float    v8f;
typedef unsigned int v4u __attribute__((ext_vector_type(4)));
typedef int          v8i __attribute__((ext_vector_type(8)));
typedef int          v4i __attribute__((ext_vector_type(4)));

#define NN 8192
#define EPS_INV 10.0f   // 1/0.1

// ---------------------------------------------------------------------------
// LDS byte offset of a generic pointer to a __shared__ object
// (addrspacecast generic->LDS, then ptrtoint gives the 32-bit LDS address)
// ---------------------------------------------------------------------------
__device__ static inline unsigned int lds_byte_off(const void* p)
{
    return (unsigned int)(unsigned long long)
        (__attribute__((address_space(3))) const void*)p;
}

// ---------------------------------------------------------------------------
// TDM: DMA a contiguous run of `nhalves` fp16 elements from global into LDS.
// Descriptor built per CDNA5 ISA D# tables (group0/1; groups 2/3 zero => <=2D).
// Issued wave-uniform (EXEC is ignored by tensor ops); caller must
// s_wait_tensorcnt before consuming.
// ---------------------------------------------------------------------------
__device__ static inline void tdm_load_halves(unsigned int lds_addr,
                                              const void* gsrc,
                                              unsigned int nhalves)
{
    unsigned long long ga = (unsigned long long)gsrc;
    v4u g0;
    g0[0] = 1u;                                        // count=1 (valid, user mode)
    g0[1] = lds_addr;                                  // LDS byte address
    g0[2] = (unsigned int)(ga & 0xffffffffull);        // global_addr[31:0]
    g0[3] = (unsigned int)((ga >> 32) & 0x1ffffffull)  // global_addr[56:32]
          | (2u << 30);                                // type=2 ("image")
    v8i g1;
    g1[0] = (int)(1u << 16);                           // data_size=1 -> 2 bytes
    g1[1] = (int)((nhalves & 0xffffu) << 16);          // tensor_dim0[15:0]
    g1[2] = (int)(((nhalves >> 16) & 0xffffu)          // tensor_dim0[31:16]
          | (1u << 16));                               // tensor_dim1 = 1
    g1[3] = (int)((nhalves & 0xffffu) << 16);          // tile_dim0 = nhalves
    g1[4] = 1;                                         // tile_dim1=1, tile_dim2=0
    g1[5] = (int)nhalves;                              // tensor_dim0_stride[31:0]
    g1[6] = 0;                                         // stride hi, dim1_stride lo
    g1[7] = 0;
    v4i g2 = {0, 0, 0, 0};
    v4i g3 = {0, 0, 0, 0};
#if defined(__clang_major__) && (__clang_major__ >= 23)
    v8i g4 = {0, 0, 0, 0, 0, 0, 0, 0};
    __builtin_amdgcn_tensor_load_to_lds(g0, g1, g2, g3, g4, 0);
#else
    __builtin_amdgcn_tensor_load_to_lds(g0, g1, g2, g3, 0);
#endif
}

// ---------------------------------------------------------------------------
// prep: Y = X @ W^T (f16 out), Yn = row squared norms; optionally XT = X^T f16
// block = 256 threads -> 4 rows, 64 dims each
// ---------------------------------------------------------------------------
__global__ __launch_bounds__(256) void prep_kernel(
    const float* __restrict__ X, const float* __restrict__ W,
    _Float16* __restrict__ Xh, float* __restrict__ Xn,
    _Float16* __restrict__ XT, int nrows)
{
    __shared__ float wl[64 * 64];
    __shared__ float xl[256];
    __shared__ float wavesum[8];
    int tid = threadIdx.x;
    for (int k = tid; k < 64 * 64; k += 256) wl[k] = W[k];
    int row0 = blockIdx.x * 4;
    xl[tid] = X[(size_t)row0 * 64 + tid];
    __syncthreads();
    int d = tid & 63, r = tid >> 6;
    int i = row0 + r;
    float val = 0.f;
#pragma unroll
    for (int k = 0; k < 64; ++k) val += xl[r * 64 + k] * wl[d * 64 + k];
    Xh[(size_t)i * 64 + d] = (_Float16)val;
    if (XT) XT[(size_t)d * nrows + i] = (_Float16)xl[r * 64 + d];
    float p = val * val;
#pragma unroll
    for (int off = 16; off; off >>= 1) p += __shfl_xor(p, off, 32);
    if ((tid & 31) == 0) wavesum[tid >> 5] = p;
    __syncthreads();
    if (d == 0) Xn[i] = wavesum[r * 2] + wavesum[r * 2 + 1];
}

// ---------------------------------------------------------------------------
// build K: per block one 128x128 tile; tiles staged via Tensor Data Mover;
// WMMA f16 (K-dim 64 = 2 x 32); C_ij = max(sn_i + tn_j - 2 s.t, 0);
// K = exp(-C/eps) -> f16 global
// ---------------------------------------------------------------------------
__global__ __launch_bounds__(256) void build_k_kernel(
    const _Float16* __restrict__ Sh, const _Float16* __restrict__ Th,
    const float* __restrict__ Sn, const float* __restrict__ Tn,
    _Float16* __restrict__ K)
{
    __shared__ __align__(16) _Float16 ls[128 * 64];
    __shared__ __align__(16) _Float16 lt[128 * 64];
    __shared__ float lsn[128];
    __shared__ float ltn[128];
    int tid = threadIdx.x;
    int tI = blockIdx.y, tJ = blockIdx.x;
    int lane = tid & 31, wave = tid >> 5;

    if (tid < 128) { lsn[tid] = Sn[tI * 128 + tid]; ltn[tid] = Tn[tJ * 128 + tid]; }
    // stage both 16KB tiles with the TDM (wave 0 issues; tensor ops ignore EXEC)
    if (wave == 0) {
        tdm_load_halves(lds_byte_off(ls), Sh + (size_t)tI * 128 * 64, 128 * 64);
        tdm_load_halves(lds_byte_off(lt), Th + (size_t)tJ * 128 * 64, 128 * 64);
        __builtin_amdgcn_s_wait_tensorcnt(0);
    }
    __syncthreads();

    int M0 = lane & 15, half = lane >> 4;
    int rowoff = wave * 16;

    // A fragments: 16x32 f16 layout (lanes 0-15: K e<8 -> e, e>=8 -> e+8; +8 upper half)
    v16h a[2];
#pragma unroll
    for (int kb = 0; kb < 2; ++kb) {
        const h8* p0 = (const h8*)(ls + (rowoff + M0) * 64 + kb * 32 + half * 8);
        const h8* p1 = (const h8*)(ls + (rowoff + M0) * 64 + kb * 32 + 16 + half * 8);
        h8 lo = p0[0], hi = p1[0];
#pragma unroll
        for (int e = 0; e < 8; ++e) { a[kb][e] = lo[e]; a[kb][8 + e] = hi[e]; }
    }

    for (int ct = 0; ct < 8; ++ct) {
        v8f c = {};
#pragma unroll
        for (int kb = 0; kb < 2; ++kb) {
            // B fragment: 32x16; lane n=M0, K e<8 -> half*8+e, e>=8 -> 16+half*8+(e-8)
            const h8* q0 = (const h8*)(lt + (ct * 16 + M0) * 64 + kb * 32 + half * 8);
            h8 lo = q0[0], hi = q0[2];
            v16h b;
#pragma unroll
            for (int e = 0; e < 8; ++e) { b[e] = lo[e]; b[8 + e] = hi[e]; }
            c = __builtin_amdgcn_wmma_f32_16x16x32_f16(false, a[kb], false, b,
                                                       (short)0, c, false, false);
        }
        int nloc = ct * 16 + M0;
        size_t gj = (size_t)tJ * 128 + nloc;
        float tn = ltn[nloc];
#pragma unroll
        for (int r = 0; r < 8; ++r) {
            int mloc = rowoff + r + half * 8;
            float cc = lsn[mloc] + tn - 2.0f * c[r];
            cc = fmaxf(cc, 0.0f);
            float kv = __expf(-cc * EPS_INV);
            size_t gi = (size_t)tI * 128 + mloc;
            K[gi * NN + gj] = (_Float16)kv;
        }
    }
}

// ---------------------------------------------------------------------------
__global__ void init_v_kernel(float* __restrict__ v)
{
    v[blockIdx.x * 256 + threadIdx.x] = 1.0f / 8192.0f;
}

// u_i = (1/N) / ((K v)_i + 1e-8); one wave per row, b128 fp16 streaming
__global__ __launch_bounds__(256) void row_update_kernel(
    const _Float16* __restrict__ K, const float* __restrict__ v,
    float* __restrict__ u)
{
    int wave = threadIdx.x >> 5, lane = threadIdx.x & 31;
    int row = blockIdx.x * 8 + wave;
    const _Float16* kr = K + (size_t)row * NN;
    float acc = 0.f;
    for (int s = 0; s < 32; ++s) {
        int j = s * 256 + lane * 8;
        __builtin_prefetch(kr + j + 512, 0, 1);   // two iterations (512 halves) ahead
        h8 kk = *(const h8*)(kr + j);
        float4 v0 = *(const float4*)(v + j);
        float4 v1 = *(const float4*)(v + j + 4);
        acc += (float)kk[0] * v0.x + (float)kk[1] * v0.y + (float)kk[2] * v0.z +
               (float)kk[3] * v0.w + (float)kk[4] * v1.x + (float)kk[5] * v1.y +
               (float)kk[6] * v1.z + (float)kk[7] * v1.w;
    }
#pragma unroll
    for (int off = 16; off; off >>= 1) acc += __shfl_xor(acc, off, 32);
    if (lane == 0) u[row] = (1.0f / 8192.0f) / (acc + 1e-8f);
}

// partial (K^T u): block = 2048 cols x 128 rows, coalesced row-major reads
__global__ __launch_bounds__(256) void col_partial_kernel(
    const _Float16* __restrict__ K, const float* __restrict__ u,
    float* __restrict__ part)
{
    int col = blockIdx.x * 2048 + threadIdx.x * 8;
    int r0 = blockIdx.y * 128;
    float acc[8] = {};
    for (int r = 0; r < 128; ++r) {
        float ui = u[r0 + r];
        const _Float16* kp = K + (size_t)(r0 + r) * NN + col;
        __builtin_prefetch(kp + 8 * NN, 0, 1);    // eight rows ahead
        h8 kk = *(const h8*)kp;
#pragma unroll
        for (int e = 0; e < 8; ++e) acc[e] += (float)kk[e] * ui;
    }
    float4* p = (float4*)(part + (size_t)blockIdx.y * NN + col);
    p[0] = make_float4(acc[0], acc[1], acc[2], acc[3]);
    p[1] = make_float4(acc[4], acc[5], acc[6], acc[7]);
}

__global__ void col_finish_kernel(const float* __restrict__ part,
                                  float* __restrict__ v)
{
    int j = blockIdx.x * 256 + threadIdx.x;
    float s = 0.f;
    for (int c = 0; c < 64; ++c) s += part[(size_t)c * NN + j];
    v[j] = (1.0f / 8192.0f) / (s + 1e-8f);
}

// ---------------------------------------------------------------------------
// final: transported = T_norm @ target via WMMA with A = K * (v*M) (scale
// cancels in normalization); row sums via B=ones WMMA; wdist via ln(K).
// One wave owns 16 rows; 4 d-tiles + 1 rowsum tile accumulated over 256 chunks.
// ---------------------------------------------------------------------------
__global__ __launch_bounds__(256) void final_kernel(
    const _Float16* __restrict__ K, const _Float16* __restrict__ TT,
    const float* __restrict__ u, const float* __restrict__ v,
    float* __restrict__ out, float* __restrict__ wpart)
{
    __shared__ float wred[8];
    int lane = threadIdx.x & 31, wave = threadIdx.x >> 5;
    int M0 = lane & 15, half = lane >> 4;
    int rowbase = (blockIdx.x * 8 + wave) * 16;
    int rowA = rowbase + M0;
    const _Float16* kr = K + (size_t)rowA * NN;

    v8f acc[4] = {{}, {}, {}, {}};
    v8f racc = {};
    v16h ones;
#pragma unroll
    for (int e = 0; e < 16; ++e) ones[e] = (_Float16)1.0f;
    float wl = 0.f;

    for (int jc = 0; jc < 256; ++jc) {
        int jb = jc * 32;
        h8 k0 = *(const h8*)(kr + jb + half * 8);
        h8 k1 = *(const h8*)(kr + jb + 16 + half * 8);
        float4 va = *(const float4*)(v + jb + half * 8);
        float4 vb = *(const float4*)(v + jb + half * 8 + 4);
        float4 vc = *(const float4*)(v + jb + 16 + half * 8);
        float4 vd = *(const float4*)(v + jb + 16 + half * 8 + 4);
        float vm[16] = {va.x, va.y, va.z, va.w, vb.x, vb.y, vb.z, vb.w,
                        vc.x, vc.y, vc.z, vc.w, vd.x, vd.y, vd.z, vd.w};
        v16h a;
#pragma unroll
        for (int e = 0; e < 8; ++e) {
            float kf = (float)k0[e];
            float af = kf * (vm[e] * 8192.0f);
            a[e] = (_Float16)af;
            if (kf > 0.f) wl += af * __logf(kf);
            float kf2 = (float)k1[e];
            float af2 = kf2 * (vm[8 + e] * 8192.0f);
            a[8 + e] = (_Float16)af2;
            if (kf2 > 0.f) wl += af2 * __logf(kf2);
        }
#pragma unroll
        for (int nt = 0; nt < 4; ++nt) {
            int dd = nt * 16 + M0;
            const h8* bp = (const h8*)(TT + (size_t)dd * NN + jb + half * 8);
            h8 lo = bp[0], hi = bp[2];
            v16h b;
#pragma unroll
            for (int e = 0; e < 8; ++e) { b[e] = lo[e]; b[8 + e] = hi[e]; }
            acc[nt] = __builtin_amdgcn_wmma_f32_16x16x32_f16(false, a, false, b,
                                                             (short)0, acc[nt], false, false);
        }
        racc = __builtin_amdgcn_wmma_f32_16x16x32_f16(false, a, false, ones,
                                                      (short)0, racc, false, false);
    }

    // transported rows
#pragma unroll
    for (int r = 0; r < 8; ++r) {
        int m = r + half * 8;
        int row = rowbase + m;
        float ui = u[row];
        float denom = ui * racc[r] * (1.0f / 8192.0f) + 1e-8f;
        float scale = ui * (1.0f / 8192.0f) / denom;
#pragma unroll
        for (int nt = 0; nt < 4; ++nt)
            out[(size_t)row * 64 + nt * 16 + M0] = acc[nt][r] * scale;
    }

    // wasserstein: sum u_i * (K v)_ij * C_ij, C = -eps ln K; fixed-order reduce
    float wterm = -(0.1f / 8192.0f) * u[rowA] * wl;
#pragma unroll
    for (int off = 16; off; off >>= 1) wterm += __shfl_xor(wterm, off, 32);
    if (lane == 0) wred[wave] = wterm;
    __syncthreads();
    if (threadIdx.x == 0) {
        float s = 0.f;
#pragma unroll
        for (int w = 0; w < 8; ++w) s += wred[w];
        wpart[blockIdx.x] = s;
    }
}

__global__ void wsum_kernel(const float* __restrict__ wpart, float* __restrict__ out)
{
    if (threadIdx.x == 0) {
        float s = 0.f;
        for (int c = 0; c < 64; ++c) s += wpart[c];
        out[(size_t)8192 * 64] = s;
    }
}

// ---------------------------------------------------------------------------
extern "C" void kernel_launch(void* const* d_in, const int* in_sizes, int n_in,
                              void* d_out, int out_size, void* d_ws, size_t ws_size,
                              hipStream_t stream)
{
    (void)in_sizes; (void)n_in; (void)out_size; (void)ws_size;
    const float* src = (const float*)d_in[0];
    const float* tgt = (const float*)d_in[1];
    const float* W   = (const float*)d_in[2];
    float* out = (float*)d_out;

    char* ws = (char*)d_ws;
    size_t off = 0;
    _Float16* Kg = (_Float16*)(ws + off); off += (size_t)NN * NN * 2;     // 128 MB, fits L2
    _Float16* Sh = (_Float16*)(ws + off); off += (size_t)NN * 64 * 2;
    _Float16* Th = (_Float16*)(ws + off); off += (size_t)NN * 64 * 2;
    _Float16* TT = (_Float16*)(ws + off); off += (size_t)NN * 64 * 2;     // target^T f16
    float* Sn    = (float*)(ws + off);    off += (size_t)NN * 4;
    float* Tn    = (float*)(ws + off);    off += (size_t)NN * 4;
    float* u     = (float*)(ws + off);    off += (size_t)NN * 4;
    float* v     = (float*)(ws + off);    off += (size_t)NN * 4;
    float* part  = (float*)(ws + off);    off += (size_t)64 * NN * 4;
    float* wpart = (float*)(ws + off);    off += 64 * 4;

    prep_kernel<<<2048, 256, 0, stream>>>(src, W, Sh, Sn, nullptr, NN);
    prep_kernel<<<2048, 256, 0, stream>>>(tgt, W, Th, Tn, TT, NN);
    build_k_kernel<<<dim3(64, 64), 256, 0, stream>>>(Sh, Th, Sn, Tn, Kg);
    init_v_kernel<<<32, 256, 0, stream>>>(v);
    for (int it = 0; it < 20; ++it) {
        row_update_kernel<<<1024, 256, 0, stream>>>(Kg, v, u);
        col_partial_kernel<<<dim3(4, 64), 256, 0, stream>>>(Kg, u, part);
        col_finish_kernel<<<32, 256, 0, stream>>>(part, v);
    }
    final_kernel<<<64, 256, 0, stream>>>(Kg, TT, u, v, out, wpart);
    wsum_kernel<<<1, 64, 0, stream>>>(wpart, out);
}
